// ProbSparseAttention_57200374448659
// MI455X (gfx1250) — compile-verified, matrix operand order checked by reading, not run on hardware
//
#include <hip/hip_runtime.h>
#include <hip/hip_bf16.h>

// ---------------- CDNA5 (gfx1250) types ----------------
typedef __attribute__((ext_vector_type(16))) __bf16 v16bf;
typedef __attribute__((ext_vector_type(8)))  float  v8f;
typedef unsigned int u32x4 __attribute__((ext_vector_type(4)));
typedef int          i32x4 __attribute__((ext_vector_type(4)));
typedef int          i32x8 __attribute__((ext_vector_type(8)));

#define D_MODEL 512
#define N_HEADS 8
#define HEAD_DIM 64
#define SEQ_LEN 2048
#define BATCH 2
#define NTOK (BATCH * SEQ_LEN)   // 4096
#define K_TOP 1024               // TOPK_RATIO * L

#if defined(__has_builtin)
#  if __has_builtin(__builtin_amdgcn_tensor_load_to_lds)
#    define HAVE_TDM 1
#  endif
#  if __has_builtin(__builtin_amdgcn_s_wait_tensorcnt)
#    define HAVE_TENSORCNT 1
#  endif
#endif

// ---------------- helpers ----------------
static __device__ __forceinline__ unsigned short f2bf(float f) {
    unsigned int u = __builtin_bit_cast(unsigned int, f);
    u += 0x7FFFu + ((u >> 16) & 1u);           // round-to-nearest-even
    return (unsigned short)(u >> 16);
}

// monotone order-preserving float -> uint key
static __device__ __forceinline__ unsigned int fkey(float f) {
    unsigned int u = __builtin_bit_cast(unsigned int, f);
    return (u & 0x80000000u) ? ~u : (u | 0x80000000u);
}

// 16x32 bf16 WMMA A-fragment (or 32x16 B-fragment of an operand that is
// row-major by its first index): lane holds row (lane&15); K-half by lane>>4.
// Two 16-byte loads per lane (global_load_b128 or ds_load_b128).
static __device__ __forceinline__ v16bf load_frag(const unsigned short* __restrict__ base,
                                                  int stride, int lane) {
    const int row = lane & 15;
    const int k0  = (lane >> 4) * 8;
    const unsigned short* p = base + (size_t)row * stride + k0;
    union { v16bf bf; uint4 q[2]; } u;
    u.q[0] = *(const uint4*)(p);
    u.q[1] = *(const uint4*)(p + 16);
    return u.bf;
}

static __device__ __forceinline__ v8f wmma_bf16(v16bf a, v16bf b, v8f c) {
    return __builtin_amdgcn_wmma_f32_16x16x32_bf16(false, a, false, b, (short)0, c, false, false);
}

static __device__ __forceinline__ float wred_max(float v) {
    #pragma unroll
    for (int off = 16; off; off >>= 1) v = fmaxf(v, __shfl_xor(v, off, 32));
    return v;
}
static __device__ __forceinline__ int wred_addi(int v) {
    #pragma unroll
    for (int off = 16; off; off >>= 1) v += __shfl_xor(v, off, 32);
    return v;
}
static __device__ __forceinline__ float wred_addf(float v) {
    #pragma unroll
    for (int off = 16; off; off >>= 1) v += __shfl_xor(v, off, 32);
    return v;
}

// ---- Tensor Data Mover: 2-D bf16 tile (tile_d0 x tile_d1 rows) -> LDS ----
// D# per CDNA5 ISA: group0 {count=1, lds_addr,
// global_addr, type=2}; group1 {data_size=2B, tensor dims, tile dims, stride}.
static __device__ __forceinline__ void tdm_load_tile_2d(unsigned lds_off, const void* gptr,
                                                        unsigned tensor_d0, unsigned tensor_d1,
                                                        unsigned tile_d0, unsigned tile_d1,
                                                        unsigned stride0) {
#ifdef HAVE_TDM
    const unsigned long long ga = (unsigned long long)gptr;
    u32x4 g0;
    g0[0] = 1u;                                                  // count=1, user mode
    g0[1] = lds_off;                                             // lds_addr (bytes)
    g0[2] = (unsigned)(ga & 0xFFFFFFFFull);                      // global_addr[31:0]
    g0[3] = (unsigned)((ga >> 32) & 0x01FFFFFFull) | (2u << 30); // addr[56:32] | type=2
    i32x8 g1;
    g1[0] = (int)(1u << 16);                                     // data_size=1 (2 bytes)
    g1[1] = (int)((tensor_d0 & 0xFFFFu) << 16);                  // tensor_dim0[15:0]
    g1[2] = (int)(((tensor_d0 >> 16) & 0xFFFFu) | ((tensor_d1 & 0xFFFFu) << 16));
    g1[3] = (int)(((tensor_d1 >> 16) & 0xFFFFu) | ((tile_d0 & 0xFFFFu) << 16));
    g1[4] = (int)(tile_d1 & 0xFFFFu);                            // tile_dim1 (tile_dim2=0)
    g1[5] = (int)stride0;                                        // tensor_dim0_stride lo
    g1[6] = 0;                                                   // stride0 hi | stride1 lo
    g1[7] = 0;
    i32x4 z4 = {0, 0, 0, 0};
#if __clang_major__ >= 23
    i32x8 z8 = {0, 0, 0, 0, 0, 0, 0, 0};
    __builtin_amdgcn_tensor_load_to_lds(g0, g1, z4, z4, z8, 0);
#else
    __builtin_amdgcn_tensor_load_to_lds(g0, g1, z4, z4, 0);
#endif
#ifdef HAVE_TENSORCNT
    __builtin_amdgcn_s_wait_tensorcnt((short)0);
#endif
#else
    (void)lds_off; (void)gptr; (void)tensor_d0; (void)tensor_d1;
    (void)tile_d0; (void)tile_d1; (void)stride0;
#endif
}

// ---------------- kernel 0: f32 -> bf16 convert ----------------
__global__ void cvt_bf16_kernel(const float* __restrict__ src,
                                unsigned short* __restrict__ dst, int n) {
    int i = blockIdx.x * blockDim.x + threadIdx.x;
    int stride = gridDim.x * blockDim.x;
    for (; i < n; i += stride) dst[i] = f2bf(src[i]);
}

// ---- shared GEMM body: per-block W tile (64x512 bf16 = 64KB) staged in LDS
// by the TDM, then every wave reads B-fragments via ds_load_b128. ----
static __device__ __forceinline__ void stage_w_tile(unsigned short* wtile,
                                                    const unsigned short* W, int nbase) {
#ifdef HAVE_TDM
    if (threadIdx.x < 32) {   // one wave issues the DMA + waits TENSORcnt
        tdm_load_tile_2d((unsigned)(unsigned long long)(uintptr_t)wtile,
                         W + (size_t)nbase * D_MODEL,
                         D_MODEL, 64, D_MODEL, 64, D_MODEL);
    }
#else
    // fallback: cooperative vector copy
    const uint4* src = (const uint4*)(W + (size_t)nbase * D_MODEL);
    uint4* dst = (uint4*)wtile;
    for (int i = threadIdx.x; i < 64 * D_MODEL / 8; i += blockDim.x) dst[i] = src[i];
#endif
    __syncthreads();
}

// ---------------- kernel 1: fused Q/K/V^T projection ----------------
// C[token, j] = sum_d xb[token,d] * W[j,d] + bias[j]; per-wave 16x64 tile.
// blockIdx.z selects Q / K / V; V is written transposed per head: [b,h,d,pos].
__global__ void proj_qkv_kernel(const unsigned short* __restrict__ xb,
                                const unsigned short* __restrict__ wq,
                                const unsigned short* __restrict__ wk,
                                const unsigned short* __restrict__ wv,
                                const float* __restrict__ bq,
                                const float* __restrict__ bk,
                                const float* __restrict__ bv,
                                unsigned short* __restrict__ qb,
                                unsigned short* __restrict__ kb,
                                unsigned short* __restrict__ vtb) {
    extern __shared__ unsigned short wtile[];        // 64 x 512 bf16
    const int lane = threadIdx.x & 31;
    const int wave = threadIdx.x >> 5;
    const int mbase = (blockIdx.x * 8 + wave) * 16;  // token base
    const int nbase = blockIdx.y * 64;               // output-feature base
    const int which = blockIdx.z;

    const unsigned short* W = (which == 0) ? wq : (which == 1) ? wk : wv;
    const float* bias       = (which == 0) ? bq : (which == 1) ? bk : bv;

    stage_w_tile(wtile, W, nbase);

    v8f acc[4] = {};
    for (int kk = 0; kk < D_MODEL; kk += 32) {
        __builtin_prefetch(xb + (size_t)mbase * D_MODEL + kk + 128, 0, 1);
        v16bf a = load_frag(xb + (size_t)mbase * D_MODEL + kk, D_MODEL, lane);
        #pragma unroll
        for (int n = 0; n < 4; n++) {
            v16bf b = load_frag(wtile + (size_t)(n * 16) * D_MODEL + kk, D_MODEL, lane);
            acc[n] = wmma_bf16(a, b, acc[n]);
        }
    }

    const int colL = lane & 15;
    const int rowOff = (lane >> 4) * 8;
    #pragma unroll
    for (int n = 0; n < 4; n++) {
        const int j = nbase + n * 16 + colL;
        const float bval = bias[j];
        const int h = j >> 6, d = j & 63;
        #pragma unroll
        for (int r = 0; r < 8; r++) {
            const int token = mbase + rowOff + r;
            const int bidx = token >> 11, pos = token & (SEQ_LEN - 1);
            const unsigned short bf = f2bf(acc[n][r] + bval);
            const size_t head = (size_t)(bidx * N_HEADS + h);
            if (which == 0)      qb [(head * SEQ_LEN + pos) * HEAD_DIM + d] = bf;
            else if (which == 1) kb [(head * SEQ_LEN + pos) * HEAD_DIM + d] = bf;
            else                 vtb[(head * HEAD_DIM + d) * SEQ_LEN + pos] = bf;
        }
    }
}

// ---------------- kernel 2: scores + exact top-k + softmax + P*V ----------------
// 4 waves (128 threads) per 16 query rows; 16x2048 f32 scores in LDS (128 KB of
// the WGP's 320 KB). Waves split key tiles in the score phase, split the 16 rows
// in the stats phase, and each wave owns one 16-wide Dh slice in the P*V phase.
// Threshold = exact 1024-th largest per row via 32-step radix search on IEEE keys.
__global__ void attn_topk_kernel(const unsigned short* __restrict__ qb,
                                 const unsigned short* __restrict__ kb,
                                 const unsigned short* __restrict__ vtb,
                                 const float* __restrict__ mask,
                                 unsigned short* __restrict__ ob) {
    extern __shared__ float smem[];                 // 16*2048 scores + 48 params
    const int lane = threadIdx.x & 31;
    const int wave = threadIdx.x >> 5;              // 0..3
    const int qbase = blockIdx.x * 16;
    const int h = blockIdx.y, b = blockIdx.z;
    const size_t headOff = (size_t)(b * N_HEADS + h) * SEQ_LEN * HEAD_DIM;
    const unsigned short* Qh = qb + headOff;
    const unsigned short* Kh = kb + headOff;
    const unsigned short* Vh = vtb + headOff;       // [d][pos]

    const int colL = lane & 15;
    const int rowOff = (lane >> 4) * 8;
    const float scale = 0.125f;                     // 1/sqrt(64)

    // Q fragments (K-chunks 0 and 32 of Dh=64), resident for the whole block
    const v16bf qa0 = load_frag(Qh + (size_t)qbase * HEAD_DIM +  0, HEAD_DIM, lane);
    const v16bf qa1 = load_frag(Qh + (size_t)qbase * HEAD_DIM + 32, HEAD_DIM, lane);

    // ---- scores: each wave handles 32 of the 128 key tiles (2 WMMA each) ----
    for (int kt = wave * 32; kt < wave * 32 + 32; kt++) {
        const int kb16 = kt * 16;
        v16bf b0 = load_frag(Kh + (size_t)kb16 * HEAD_DIM +  0, HEAD_DIM, lane);
        v16bf b1 = load_frag(Kh + (size_t)kb16 * HEAD_DIM + 32, HEAD_DIM, lane);
        v8f acc = {};
        acc = wmma_bf16(qa0, b0, acc);
        acc = wmma_bf16(qa1, b1, acc);
        const int col = kb16 + colL;
        #pragma unroll
        for (int r = 0; r < 8; r++) {
            const int qrow = rowOff + r;
            const float mk = mask[(size_t)(qbase + qrow) * SEQ_LEN + col];
            smem[qrow * SEQ_LEN + col] = acc[r] * scale + mk;
        }
    }
    __syncthreads();

    // ---- per-row stats: wave w owns rows 4w..4w+3 ----
    float* prm = smem + 16 * SEQ_LEN;               // [0..15]=max [16..31]=keyT [32..47]=denom
    for (int r = wave * 4; r < wave * 4 + 4; r++) {
        const float* row = smem + r * SEQ_LEN;
        float lm = -3.0e38f;
        for (int i = lane; i < SEQ_LEN; i += 32) lm = fmaxf(lm, row[i]);
        lm = wred_max(lm);

        unsigned int acck = 0u;                     // radix binary search: largest T
        #pragma unroll 1                            // with count(key>=T) >= K_TOP
        for (int bit = 31; bit >= 0; bit--) {
            const unsigned int cand = acck | (1u << bit);
            int cnt = 0;
            for (int i = lane; i < SEQ_LEN; i += 32) cnt += (fkey(row[i]) >= cand) ? 1 : 0;
            cnt = wred_addi(cnt);
            if (cnt >= K_TOP) acck = cand;
        }

        float ls = 0.f;
        for (int i = lane; i < SEQ_LEN; i += 32) {
            const float s = row[i];
            if (fkey(s) >= acck) ls += __expf(s - lm);
        }
        ls = wred_addf(ls);
        if (lane == 0) {
            prm[r] = lm;
            prm[16 + r] = __builtin_bit_cast(float, acck);
            prm[32 + r] = ls;
        }
    }
    __syncthreads();

    // per-lane row params for A-fragment building (lane's A-row = lane&15)
    const float mRow = prm[colL];
    const unsigned int keyT = __builtin_bit_cast(unsigned int, prm[16 + colL]);
    const float dInv = 1.0f / prm[32 + colL];

    // ---- P * V : wave w owns Dh slice [16w,16w+16); 64 K-chunks of 32 keys ----
    v8f oacc = {};
    for (int kc = 0; kc < SEQ_LEN / 32; kc++) {
        const int kb32 = kc * 32;
        union { v16bf bf; unsigned short us[16]; } ua;
        const float* rowp = smem + (size_t)colL * SEQ_LEN + kb32 + rowOff;
        #pragma unroll
        for (int i = 0; i < 8; i++) {
            const float s0 = rowp[i];
            const float s1 = rowp[16 + i];
            const float p0 = (fkey(s0) >= keyT) ? __expf(s0 - mRow) * dInv : 0.f;
            const float p1 = (fkey(s1) >= keyT) ? __expf(s1 - mRow) * dInv : 0.f;
            ua.us[i]     = f2bf(p0);
            ua.us[8 + i] = f2bf(p1);
        }
        v16bf vb = load_frag(Vh + (size_t)(wave * 16) * SEQ_LEN + kb32, SEQ_LEN, lane);
        oacc = wmma_bf16(ua.bf, vb, oacc);
    }

    // store attention output as bf16 in [token, h*64+d] layout for the O-proj GEMM
    const int d = wave * 16 + colL;
    #pragma unroll
    for (int r = 0; r < 8; r++) {
        const int pos = qbase + rowOff + r;
        const size_t token = (size_t)b * SEQ_LEN + pos;
        ob[token * D_MODEL + h * HEAD_DIM + d] = f2bf(oacc[r]);
    }
}

// ---------------- kernel 3: output projection (f32 result) ----------------
__global__ void oproj_kernel(const unsigned short* __restrict__ obuf,
                             const unsigned short* __restrict__ wo,
                             const float* __restrict__ bo,
                             float* __restrict__ out) {
    extern __shared__ unsigned short wtile[];        // 64 x 512 bf16
    const int lane = threadIdx.x & 31;
    const int wave = threadIdx.x >> 5;
    const int mbase = (blockIdx.x * 8 + wave) * 16;
    const int nbase = blockIdx.y * 64;

    stage_w_tile(wtile, wo, nbase);

    v8f acc[4] = {};
    for (int kk = 0; kk < D_MODEL; kk += 32) {
        __builtin_prefetch(obuf + (size_t)mbase * D_MODEL + kk + 128, 0, 1);
        v16bf a = load_frag(obuf + (size_t)mbase * D_MODEL + kk, D_MODEL, lane);
        #pragma unroll
        for (int n = 0; n < 4; n++) {
            v16bf b = load_frag(wtile + (size_t)(n * 16) * D_MODEL + kk, D_MODEL, lane);
            acc[n] = wmma_bf16(a, b, acc[n]);
        }
    }
    const int colL = lane & 15;
    const int rowOff = (lane >> 4) * 8;
    #pragma unroll
    for (int n = 0; n < 4; n++) {
        const int j = nbase + n * 16 + colL;
        const float bval = bo[j];
        #pragma unroll
        for (int r = 0; r < 8; r++) {
            const int token = mbase + rowOff + r;
            out[(size_t)token * D_MODEL + j] = acc[n][r] + bval;
        }
    }
}

// ---------------- launch ----------------
extern "C" void kernel_launch(void* const* d_in, const int* in_sizes, int n_in,
                              void* d_out, int out_size, void* d_ws, size_t ws_size,
                              hipStream_t stream) {
    const float* x    = (const float*)d_in[0];
    const float* mask = (const float*)d_in[1];
    const float* Wq   = (const float*)d_in[2];
    const float* bq   = (const float*)d_in[3];
    const float* Wk   = (const float*)d_in[4];
    const float* bk   = (const float*)d_in[5];
    const float* Wv   = (const float*)d_in[6];
    const float* bv   = (const float*)d_in[7];
    const float* Wo   = (const float*)d_in[8];
    const float* bo   = (const float*)d_in[9];
    float* out = (float*)d_out;

    // workspace layout (bf16 stored as ushort)
    const size_t XN = (size_t)NTOK * D_MODEL;             // 4096*512
    const size_t WN = (size_t)D_MODEL * D_MODEL;          // 512*512
    const size_t HN = (size_t)BATCH * N_HEADS * SEQ_LEN * HEAD_DIM;
    unsigned short* ws   = (unsigned short*)d_ws;
    unsigned short* xb   = ws;
    unsigned short* wqb  = xb + XN;
    unsigned short* wkb  = wqb + WN;
    unsigned short* wvb  = wkb + WN;
    unsigned short* wob  = wvb + WN;
    unsigned short* qbuf = wob + WN;
    unsigned short* kbuf = qbuf + HN;
    unsigned short* vtbf = kbuf + HN;
    unsigned short* obuf = vtbf + HN;

    // 0) f32 -> bf16 conversions
    cvt_bf16_kernel<<<1024, 256, 0, stream>>>(x,  xb,  (int)XN);
    cvt_bf16_kernel<<<512,  256, 0, stream>>>(Wq, wqb, (int)WN);
    cvt_bf16_kernel<<<512,  256, 0, stream>>>(Wk, wkb, (int)WN);
    cvt_bf16_kernel<<<512,  256, 0, stream>>>(Wv, wvb, (int)WN);
    cvt_bf16_kernel<<<512,  256, 0, stream>>>(Wo, wob, (int)WN);

    // 1) Q/K/V^T projections: 8 waves/block, TDM-staged 64KB W tile in LDS
    const size_t wtile_bytes = (size_t)64 * D_MODEL * sizeof(unsigned short);
    proj_qkv_kernel<<<dim3(NTOK / 128, D_MODEL / 64, 3), 256, wtile_bytes, stream>>>(
        xb, wqb, wkb, wvb, bq, bk, bv, qbuf, kbuf, vtbf);

    // 2) attention: 4 waves per 16 queries; 128KB dynamic LDS per block
    const size_t smem_bytes = (size_t)(16 * SEQ_LEN + 48) * sizeof(float);
    attn_topk_kernel<<<dim3(SEQ_LEN / 16, N_HEADS, BATCH), 128, smem_bytes, stream>>>(
        qbuf, kbuf, vtbf, mask, obuf);

    // 3) output projection -> f32 d_out
    oproj_kernel<<<dim3(NTOK / 128, D_MODEL / 64), 256, wtile_bytes, stream>>>(obuf, wob, bo, out);
}
